// QuantumProcessingUnit_45140106281023
// MI455X (gfx1250) — matrix operand order repaired and checked by reading, not compiled, not run on hardware
//
#include <hip/hip_runtime.h>
#include <hip/hip_bf16.h>
#include <math.h>

typedef __attribute__((ext_vector_type(16))) _Float16 v16h;
typedef __attribute__((ext_vector_type(8)))  float    v8f;

#define ACT_GELU 1
#define ACT_TANH 2

__device__ __forceinline__ float gelu_exact(float v) {
  return 0.5f * v * (1.0f + erff(v * 0.70710678118654752440f));
}

// ---------------------------------------------------------------------------
// WMMA GEMM:  C[M,N] = act(A[M,K] @ W[N,K]^T + bias)
// One wave32 computes one 16x16 output tile with V_WMMA_F32_16X16X32_F16,
// using a hi/lo f16 split of the f32 operands for ~f32 accuracy:
//   a*b ~= ah*bh + ah*bl + al*bh   (drop al*bl ~ 2^-22 relative)
// Fragment layouts per CDNA5 ISA 7.12.2:
//   A (16x32 f16): lane&15 = M row; lanes<16 hold K {0..7,16..23},
//                  lanes>=16 hold K {8..15,24..31} (element e -> e<8?e:e+8)
//   B (32x16 f16): lane&15 = N col; lanes<16 hold K {0..7,16..23}, lanes>=16 K {8..15,24..31}
//   C/D f32:       VGPR r -> M = r + 8*(lane>=16), N = lane&15
// ---------------------------------------------------------------------------
template <int ACT>
__global__ void __launch_bounds__(256) gemm_wmma_kernel(
    const float* __restrict__ A, const float* __restrict__ W,
    const float* __restrict__ bias, float* __restrict__ C,
    int M, int N, int K, int ntilesN) {
  const int wave = (int)((blockIdx.x * blockDim.x + threadIdx.x) >> 5);
  const int lane = (int)(threadIdx.x & 31);
  const int tileM = (wave / ntilesN) * 16;
  const int tileN = (wave % ntilesN) * 16;
  if (tileM >= M) return;

  const int halfSel = (lane & 16) ? 1 : 0;   // which half-wave this lane is in
  const int l15 = lane & 15;
  const int rowA = tileM + l15;
  const int colB = tileN + l15;
  const bool colOK = (colB < N);

  v8f acc = {0.f, 0.f, 0.f, 0.f, 0.f, 0.f, 0.f, 0.f};

  for (int kb = 0; kb < K; kb += 32) {
    __builtin_prefetch(&A[rowA * K + kb + 32], 0, 3);  // global_prefetch_b8
    v16h ahi, alo, bhi, blo;
#pragma unroll
    for (int e = 0; e < 16; ++e) {
      // A-fragment element -> K index
      const int kA = kb + (halfSel ? 8 : 0) + (e < 8 ? e : e + 8);
      const float av = A[rowA * K + kA];
      const _Float16 ah = (_Float16)av;
      ahi[e] = ah;
      alo[e] = (_Float16)(av - (float)ah);
      // B-fragment element -> K index (B[k][n] = W[n][k])
      const int kB = kb + (halfSel ? 8 : 0) + (e < 8 ? e : e + 8);
      const float bv = colOK ? W[colB * K + kB] : 0.0f;
      const _Float16 bh = (_Float16)bv;
      bhi[e] = bh;
      blo[e] = (_Float16)(bv - (float)bh);
    }
    acc = __builtin_amdgcn_wmma_f32_16x16x32_f16(false, ahi, false, bhi,
                                                 (short)0, acc, false, false);
    acc = __builtin_amdgcn_wmma_f32_16x16x32_f16(false, ahi, false, blo,
                                                 (short)0, acc, false, false);
    acc = __builtin_amdgcn_wmma_f32_16x16x32_f16(false, alo, false, bhi,
                                                 (short)0, acc, false, false);
  }

#pragma unroll
  for (int r = 0; r < 8; ++r) {
    const int m = tileM + r + halfSel * 8;
    const int n = tileN + l15;
    if (n < N) {
      float v = acc[r] + bias[n];
      if (ACT == ACT_GELU) v = gelu_exact(v);
      else if (ACT == ACT_TANH) v = tanhf(v);
      C[m * N + n] = v;
    }
  }
}

// ---------------------------------------------------------------------------
// LayerNorm over rows of H (256 x 128), in place. One wave per row.
// ---------------------------------------------------------------------------
__global__ void __launch_bounds__(256) layernorm_kernel(
    float* __restrict__ H, const float* __restrict__ g,
    const float* __restrict__ b) {
  const int row = (int)((blockIdx.x * blockDim.x + threadIdx.x) >> 5);
  const int lane = (int)(threadIdx.x & 31);
  float v[4];
  float sum = 0.f;
#pragma unroll
  for (int j = 0; j < 4; ++j) {
    v[j] = H[row * 128 + lane + 32 * j];
    sum += v[j];
  }
#pragma unroll
  for (int off = 16; off > 0; off >>= 1) sum += __shfl_xor(sum, off, 32);
  const float mu = sum * (1.0f / 128.0f);
  float var = 0.f;
#pragma unroll
  for (int j = 0; j < 4; ++j) {
    const float d = v[j] - mu;
    var += d * d;
  }
#pragma unroll
  for (int off = 16; off > 0; off >>= 1) var += __shfl_xor(var, off, 32);
  const float rstd = rsqrtf(var * (1.0f / 128.0f) + 1e-5f);
#pragma unroll
  for (int j = 0; j < 4; ++j) {
    const int col = lane + 32 * j;
    H[row * 128 + col] = (v[j] - mu) * rstd * g[col] + b[col];
  }
}

// ---------------------------------------------------------------------------
// 12-qubit state-vector simulator. One workgroup (256 thr) per batch sample,
// full state (4096 complex) resident in LDS. Qubit q <-> flat-index bit 11-q.
// ---------------------------------------------------------------------------
#define NQ 12
#define NSTATES 4096
#define NPAIRS 2048
#define QTHREADS 256

__device__ __forceinline__ void apply_ry(float* sre, float* sim, int q,
                                         float angle, int tid) {
  const unsigned bit = 1u << (11 - q);
  const float c = cosf(0.5f * angle);
  const float s = sinf(0.5f * angle);
  for (int t = tid; t < NPAIRS; t += QTHREADS) {
    const unsigned u = (unsigned)t;
    const unsigned i0 = (((u & ~(bit - 1u)) << 1) | (u & (bit - 1u)));
    const unsigned i1 = i0 | bit;
    const float r0 = sre[i0], m0 = sim[i0];
    const float r1 = sre[i1], m1 = sim[i1];
    sre[i0] = c * r0 - s * r1;  sim[i0] = c * m0 - s * m1;
    sre[i1] = s * r0 + c * r1;  sim[i1] = s * m0 + c * m1;
  }
  __syncthreads();
}

__device__ __forceinline__ void apply_u(float* sre, float* sim, int q,
                                        float u00r, float u00i, float u01r, float u01i,
                                        float u10r, float u10i, float u11r, float u11i,
                                        int tid) {
  const unsigned bit = 1u << (11 - q);
  for (int t = tid; t < NPAIRS; t += QTHREADS) {
    const unsigned u = (unsigned)t;
    const unsigned i0 = (((u & ~(bit - 1u)) << 1) | (u & (bit - 1u)));
    const unsigned i1 = i0 | bit;
    const float r0 = sre[i0], m0 = sim[i0];
    const float r1 = sre[i1], m1 = sim[i1];
    sre[i0] = u00r * r0 - u00i * m0 + u01r * r1 - u01i * m1;
    sim[i0] = u00r * m0 + u00i * r0 + u01r * m1 + u01i * r1;
    sre[i1] = u10r * r0 - u10i * m0 + u11r * r1 - u11i * m1;
    sim[i1] = u10r * m0 + u10i * r0 + u11r * m1 + u11i * r1;
  }
  __syncthreads();
}

__device__ __forceinline__ void apply_cnot(float* sre, float* sim, int c, int t,
                                           int tid) {
  const unsigned bc = 1u << (11 - c);
  const unsigned bt = 1u << (11 - t);
  const unsigned hi = (bc > bt) ? bc : bt;
  const unsigned lo = (bc > bt) ? bt : bc;
  for (int k = tid; k < (NSTATES / 4); k += QTHREADS) {
    unsigned x = (unsigned)k;
    x = ((x & ~(lo - 1u)) << 1) | (x & (lo - 1u));
    x = ((x & ~(hi - 1u)) << 1) | (x & (hi - 1u));
    const unsigned i0 = x | bc;        // control=1, target=0
    const unsigned i1 = i0 | bt;       // control=1, target=1
    const float r = sre[i0]; sre[i0] = sre[i1]; sre[i1] = r;
    const float m = sim[i0]; sim[i0] = sim[i1]; sim[i1] = m;
  }
  __syncthreads();
}

// Fused layer of commuting CZ(i, i+2), i in {0,2,4,6,8}: one parity-sign pass.
__device__ __forceinline__ void apply_cz_layer(float* sre, float* sim, int tid) {
  for (int i = tid; i < NSTATES; i += QTHREADS) {
    const unsigned u = (unsigned)i;
    unsigned par = 0u;
#pragma unroll
    for (int q = 0; q < 10; q += 2) {
      const unsigned b1 = 1u << (11 - q);
      const unsigned b2 = 1u << (9 - q);
      par ^= (unsigned)(((u & b1) != 0u) & ((u & b2) != 0u));
    }
    if (par) { sre[i] = -sre[i]; sim[i] = -sim[i]; }
  }
  __syncthreads();
}

__global__ void __launch_bounds__(QTHREADS) quantum_kernel(
    const float* __restrict__ xq, const float* __restrict__ qw,
    float* __restrict__ qout) {
  const int b = (int)blockIdx.x;
  const int tid = (int)threadIdx.x;

  __shared__ float sre[NSTATES];
  __shared__ float sim[NSTATES];
  __shared__ float red[24];

  for (int i = tid; i < NSTATES; i += QTHREADS) { sre[i] = 0.f; sim[i] = 0.f; }
  if (tid == 0) sre[0] = 1.0f;
  __syncthreads();

  float enc[NQ];
#pragma unroll
  for (int q = 0; q < NQ; ++q)
    enc[q] = xq[b * NQ + q] * 9.8696044010893586f;  // pi^2

  // Initial per-sample RY encoding
  for (int q = 0; q < NQ; ++q) apply_ry(sre, sim, q, enc[q], tid);

  for (int layer = 0; layer < 6; ++layer) {
    // Shared-weight rotations: U = RZ(w2) RY(w1) RZ(w0)
    for (int i = 0; i < NQ; ++i) {
      const float w0 = qw[(layer * NQ + i) * 3 + 0];
      const float w1 = qw[(layer * NQ + i) * 3 + 1];
      const float w2 = qw[(layer * NQ + i) * 3 + 2];
      const float alpha = 0.5f * (w0 + w2);
      const float beta  = 0.5f * (w0 - w2);
      const float ca = cosf(alpha), sa = sinf(alpha);
      const float cb = cosf(beta),  sb = sinf(beta);
      const float c1 = cosf(0.5f * w1), s1 = sinf(0.5f * w1);
      // u00 = c1 e^{-i a}; u01 = -s1 e^{+i b}; u10 = s1 e^{-i b}; u11 = c1 e^{+i a}
      apply_u(sre, sim, i,
              c1 * ca, -c1 * sa,
              -s1 * cb, -s1 * sb,
              s1 * cb, -s1 * sb,
              c1 * ca, c1 * sa, tid);
    }
    // CNOT ring (sequential; overlapping chain does not commute)
    for (int i = 0; i < NQ; ++i) apply_cnot(sre, sim, i, (i + 1) % NQ, tid);
    // Re-encoding RY on even layers (except last layer)
    if ((layer % 2) == 0 && layer < 5)
      for (int i = 0; i < NQ; ++i) apply_ry(sre, sim, i, enc[i] * 0.5f, tid);
    // CZ layer on odd layers
    if ((layer % 2) == 1) apply_cz_layer(sre, sim, tid);
  }

  // <Z> on qubits 0,1,2 (flat-index bits 11,10,9)
  float z0 = 0.f, z1 = 0.f, z2 = 0.f;
  for (int i = tid; i < NSTATES; i += QTHREADS) {
    const float p = sre[i] * sre[i] + sim[i] * sim[i];
    z0 += (i & 2048) ? -p : p;
    z1 += (i & 1024) ? -p : p;
    z2 += (i & 512)  ? -p : p;
  }
#pragma unroll
  for (int off = 16; off > 0; off >>= 1) {
    z0 += __shfl_xor(z0, off, 32);
    z1 += __shfl_xor(z1, off, 32);
    z2 += __shfl_xor(z2, off, 32);
  }
  const int wv = tid >> 5;
  if ((tid & 31) == 0) { red[wv] = z0; red[8 + wv] = z1; red[16 + wv] = z2; }
  __syncthreads();
  if (tid == 0) {
    float a = 0.f, c = 0.f, d = 0.f;
#pragma unroll
    for (int w = 0; w < 8; ++w) { a += red[w]; c += red[8 + w]; d += red[16 + w]; }
    qout[b * 3 + 0] = a;
    qout[b * 3 + 1] = c;
    qout[b * 3 + 2] = d;
  }
}

// ---------------------------------------------------------------------------
// Head: out = gelu(q @ W4^T + b4) @ W5^T + b5. One block (64 thr) per sample.
// ---------------------------------------------------------------------------
__global__ void __launch_bounds__(64) head_kernel(
    const float* __restrict__ Q, const float* __restrict__ W4,
    const float* __restrict__ b4, const float* __restrict__ W5,
    const float* __restrict__ b5, float* __restrict__ out) {
  const int b = (int)blockIdx.x;
  const int tid = (int)threadIdx.x;
  __shared__ float h[32];
  __shared__ float qv[3];
  if (tid < 3) qv[tid] = Q[b * 3 + tid];
  __syncthreads();
  if (tid < 32) {
    float a = b4[tid];
#pragma unroll
    for (int k = 0; k < 3; ++k) a += qv[k] * W4[tid * 3 + k];
    h[tid] = gelu_exact(a);
  }
  __syncthreads();
  float o = b5[tid];
#pragma unroll
  for (int j = 0; j < 32; ++j) o += h[j] * W5[tid * 32 + j];
  out[b * 64 + tid] = o;
}

// ---------------------------------------------------------------------------
// Launcher. Input order per setup_inputs():
//  0:x 1:W1 2:b1 3:g_ln 4:b_ln 5:W2 6:b2 7:W3 8:b3 9:q_weights 10:W4 11:b4
//  12:W5 13:b5.  Output: (256,64) f32.
// ---------------------------------------------------------------------------
extern "C" void kernel_launch(void* const* d_in, const int* in_sizes, int n_in,
                              void* d_out, int out_size, void* d_ws, size_t ws_size,
                              hipStream_t stream) {
  const float* x    = (const float*)d_in[0];
  const float* W1   = (const float*)d_in[1];
  const float* b1   = (const float*)d_in[2];
  const float* g_ln = (const float*)d_in[3];
  const float* b_ln = (const float*)d_in[4];
  const float* W2   = (const float*)d_in[5];
  const float* b2   = (const float*)d_in[6];
  const float* W3   = (const float*)d_in[7];
  const float* b3   = (const float*)d_in[8];
  const float* qw   = (const float*)d_in[9];
  const float* W4   = (const float*)d_in[10];
  const float* b4   = (const float*)d_in[11];
  const float* W5   = (const float*)d_in[12];
  const float* b5   = (const float*)d_in[13];
  float* out = (float*)d_out;

  char* ws = (char*)d_ws;
  float* H1 = (float*)(ws);                             // 256*128 f32 = 128KB
  float* H2 = (float*)(ws + 131072);                    // 256*64  f32 =  64KB
  float* XQ = (float*)(ws + 131072 + 65536);            // 256*12  f32
  float* Qz = (float*)(ws + 131072 + 65536 + 12288);    // 256*3   f32

  // GEMM1: (256x256)@(256x128) + gelu -> H1.  128 tiles, 8 waves/block.
  gemm_wmma_kernel<ACT_GELU><<<16, 256, 0, stream>>>(x, W1, b1, H1, 256, 128, 256, 8);
  // LayerNorm rows of H1 (one wave per row).
  layernorm_kernel<<<32, 256, 0, stream>>>(H1, g_ln, b_ln);
  // GEMM2: (256x128)@(128x64) + gelu -> H2.  64 tiles.
  gemm_wmma_kernel<ACT_GELU><<<8, 256, 0, stream>>>(H1, W2, b2, H2, 256, 64, 128, 4);
  // GEMM3: (256x64)@(64x12) + tanh -> XQ (N=12 padded inside tile). 16 tiles.
  gemm_wmma_kernel<ACT_TANH><<<2, 256, 0, stream>>>(H2, W3, b3, XQ, 256, 12, 64, 1);
  // Quantum circuit: one workgroup per sample, state in LDS.
  quantum_kernel<<<256, 256, 0, stream>>>(XQ, qw, Qz);
  // Head MLP.
  head_kernel<<<256, 64, 0, stream>>>(Qz, W4, b4, W5, b5, out);
}
